// Gated_DiBA_OLDA_Attention_84902913507730
// MI455X (gfx1250) — matrix operand order repaired and checked by reading, not compiled, not run on hardware
//
#include <hip/hip_runtime.h>
#include <hip/hip_bf16.h>
#include <math.h>

constexpr int kS  = 2048;
constexpr int kD  = 1024;
constexpr int kH  = 16;
constexpr int kDH = 64;
constexpr int kDL = 64;
constexpr int kHD = 1024;  // kH * kDH

typedef __bf16 bf16;
typedef __attribute__((ext_vector_type(16))) __bf16 v16bf;
typedef __attribute__((ext_vector_type(8)))  float  v8f;
typedef __attribute__((ext_vector_type(4)))  unsigned int v4u;
typedef __attribute__((ext_vector_type(8)))  int v8i;
typedef __attribute__((ext_vector_type(4)))  int v4i;

#define WMMA_BF16(a, b, c) \
  __builtin_amdgcn_wmma_f32_16x16x32_bf16(false, (a), false, (b), (short)0, (c), false, false)

// ---- Tensor Data Mover (gfx1250) --------------------------------------------
#if defined(__has_builtin)
#  if __has_builtin(__builtin_amdgcn_tensor_load_to_lds)
#    define USE_TDM 1
#  endif
#endif
#ifndef USE_TDM
#  define USE_TDM 0
#endif

#if USE_TDM
// Issue one 2D-tile DMA: global (row-major, 2-byte elems, row stride in elems)
// -> LDS at byte address lds_addr. Descriptor per CDNA5 ISA ch.8 (D# groups).
__device__ __forceinline__ void tdm_load_tile_2d(unsigned lds_addr, const void* gptr,
                                                 unsigned tile_cols, unsigned tile_rows,
                                                 unsigned tensor_cols, unsigned tensor_rows,
                                                 unsigned row_stride_elems) {
  unsigned long long ga = (unsigned long long)(size_t)gptr;
  v4u g0;
  g0[0] = 1u;                                                // count=1, user D#
  g0[1] = lds_addr;                                          // lds_addr (bytes)
  g0[2] = (unsigned)(ga & 0xffffffffu);                      // global_addr[31:0]
  g0[3] = (unsigned)((ga >> 32) & 0x1ffffffu) | (2u << 30);  // addr[56:32] | type=2
  v8i g1;
  g1[0] = (int)(1u << 16);                                   // data_size=1 (2 bytes)
  g1[1] = (int)((tensor_cols & 0xffffu) << 16);              // tensor_dim0[15:0]
  g1[2] = (int)(((tensor_cols >> 16) & 0xffffu) |            // tensor_dim0[31:16]
                ((tensor_rows & 0xffffu) << 16));            // tensor_dim1[15:0]
  g1[3] = (int)(((tensor_rows >> 16) & 0xffffu) |            // tensor_dim1[31:16]
                ((tile_cols & 0xffffu) << 16));              // tile_dim0
  g1[4] = (int)(tile_rows & 0xffffu);                        // tile_dim1 (tile_dim2=0)
  g1[5] = (int)row_stride_elems;                             // tensor_dim0_stride[31:0]
  g1[6] = 0;
  g1[7] = 0;
  v4i gz = {0, 0, 0, 0};
#if __clang_major__ >= 23
  v8i gz8 = {0, 0, 0, 0, 0, 0, 0, 0};
  __builtin_amdgcn_tensor_load_to_lds(g0, g1, gz, gz, gz8, 0);
#else
  __builtin_amdgcn_tensor_load_to_lds(g0, g1, gz, gz, 0);
#endif
}
#endif  // USE_TDM

__device__ __forceinline__ void tdm_wait0() {
#if defined(__has_builtin) && __has_builtin(__builtin_amdgcn_s_wait_tensorcnt)
  __builtin_amdgcn_s_wait_tensorcnt((short)0);
#else
  asm volatile("s_wait_tensorcnt 0x0" ::: "memory");
#endif
}
__device__ __forceinline__ void tdm_wait2() {
#if defined(__has_builtin) && __has_builtin(__builtin_amdgcn_s_wait_tensorcnt)
  __builtin_amdgcn_s_wait_tensorcnt((short)2);
#else
  asm volatile("s_wait_tensorcnt 0x2" ::: "memory");
#endif
}

// ---- WMMA fragment loaders (row-major source) --------------------------------
// A (16x32, MxK): lane m = lane&15; elems 0..7 -> K = hi*8+j, elems 8..15 -> K = 16+hi*8+j
__device__ __forceinline__ v16bf load_fragA(const bf16* base, int stride, int lane) {
  const int r = lane & 15, hi = lane >> 4;
  const bf16* p = base + (size_t)r * stride + hi * 8;
  v16bf f;
#pragma unroll
  for (int j = 0; j < 8; ++j) f[j] = p[j];
#pragma unroll
  for (int j = 0; j < 8; ++j) f[8 + j] = p[16 + j];
  return f;
}
// B (32x16, KxN) given as N-major rows (i.e. B^T row n): lane n = lane&15; elems 0..7 -> K = hi*8+j, elems 8..15 -> K = 16+hi*8+j
__device__ __forceinline__ v16bf load_fragB(const bf16* base, int stride, int lane) {
  const int n = lane & 15, hi = lane >> 4;
  const bf16* p = base + (size_t)n * stride + hi * 8;
  v16bf f;
#pragma unroll
  for (int j = 0; j < 8; ++j) f[j] = p[j];
#pragma unroll
  for (int j = 0; j < 8; ++j) f[8 + j] = p[16 + j];
  return f;
}

// ---- Generic bf16 WMMA GEMM: C[M,N] = A[MxK] * B[NxK]^T, f32 out -------------
// Block = 256 threads (8 waves), tile 128(M) x 64(N), K-step 32.
// TDM path: double-buffered LDS tiles filled by the Tensor Data Mover
// (wave 0 issues 2 descriptors per stage, pipelined on TENSORcnt).
__global__ __launch_bounds__(256) void k_gemm(const bf16* __restrict__ A,
                                              const bf16* __restrict__ B,
                                              float* __restrict__ C,
                                              int M, int N, int K) {
  __shared__ __align__(16) bf16 As[2][128 * 32];
  __shared__ __align__(16) bf16 Bs[2][64 * 32];
  const int tid = threadIdx.x;
  const int wave = tid >> 5, lane = tid & 31;
  const int m0 = blockIdx.y * 128, n0 = blockIdx.x * 64;
  v8f acc0 = {}, acc1 = {}, acc2 = {}, acc3 = {};
#if USE_TDM
  const unsigned ldsA0 = (unsigned)(size_t)&As[0][0];
  const unsigned ldsA1 = (unsigned)(size_t)&As[1][0];
  const unsigned ldsB0 = (unsigned)(size_t)&Bs[0][0];
  const unsigned ldsB1 = (unsigned)(size_t)&Bs[1][0];
  if (wave == 0) {  // prime pipeline: tiles for k0 = 0 into buffer 0
    tdm_load_tile_2d(ldsA0, &A[(size_t)m0 * K], 32, 128, (unsigned)K, (unsigned)M, (unsigned)K);
    tdm_load_tile_2d(ldsB0, &B[(size_t)n0 * K], 32, 64, (unsigned)K, (unsigned)N, (unsigned)K);
  }
#endif
  int buf = 0;
  for (int k0 = 0; k0 < K; k0 += 32) {
#if USE_TDM
    if (wave == 0) {
      if (k0 + 32 < K) {  // issue next stage into the other buffer, keep 2 in flight
        tdm_load_tile_2d(buf ? ldsA0 : ldsA1, &A[(size_t)m0 * K + k0 + 32], 32, 128,
                         (unsigned)K, (unsigned)M, (unsigned)K);
        tdm_load_tile_2d(buf ? ldsB0 : ldsB1, &B[(size_t)n0 * K + k0 + 32], 32, 64,
                         (unsigned)K, (unsigned)N, (unsigned)K);
        tdm_wait2();  // current pair landed, next pair still in flight
      } else {
        tdm_wait0();
      }
    }
    __syncthreads();
#else
#pragma unroll
    for (int it = 0; it < 2; ++it) {
      int idx = (tid + it * 256) * 8;
      int r = idx >> 5, c = idx & 31;
      *(uint4*)&As[0][idx] = *(const uint4*)&A[(size_t)(m0 + r) * K + k0 + c];
    }
    {
      int idx = tid * 8;
      int r = idx >> 5, c = idx & 31;
      *(uint4*)&Bs[0][idx] = *(const uint4*)&B[(size_t)(n0 + r) * K + k0 + c];
    }
    if (k0 + 32 < K)  // warm L2/L0 for next K tile (global_prefetch_b8)
      __builtin_prefetch(&A[(size_t)(m0 + (tid >> 1)) * K + k0 + 32], 0, 1);
    __syncthreads();
#endif
    const bf16* Asb = &As[buf][0];
    const bf16* Bsb = &Bs[buf][0];
    v16bf a  = load_fragA(Asb + wave * 16 * 32, 32, lane);
    v16bf b0 = load_fragB(Bsb + 0 * 32, 32, lane);
    v16bf b1 = load_fragB(Bsb + 16 * 32, 32, lane);
    v16bf b2 = load_fragB(Bsb + 32 * 32, 32, lane);
    v16bf b3 = load_fragB(Bsb + 48 * 32, 32, lane);
    acc0 = WMMA_BF16(a, b0, acc0);
    acc1 = WMMA_BF16(a, b1, acc1);
    acc2 = WMMA_BF16(a, b2, acc2);
    acc3 = WMMA_BF16(a, b3, acc3);
    __syncthreads();
#if USE_TDM
    buf ^= 1;
#endif
  }
  const int n = lane & 15, hi = lane >> 4;
#pragma unroll
  for (int r = 0; r < 8; ++r) {
    float* cp = C + (size_t)(m0 + wave * 16 + hi * 8 + r) * N + n0 + n;
    cp[0]  = acc0[r];
    cp[16] = acc1[r];
    cp[32] = acc2[r];
    cp[48] = acc3[r];
  }
}

// ---- small helper kernels ----------------------------------------------------
__global__ void k_f32_to_bf16(const float* __restrict__ in, bf16* __restrict__ out, int n) {
  for (int i = blockIdx.x * blockDim.x + threadIdx.x; i < n; i += gridDim.x * blockDim.x)
    out[i] = (bf16)in[i];
}

__global__ void k_wabs_scale(const float* __restrict__ w, int n, float* __restrict__ scale_out) {
  __shared__ float red[256];
  float s = 0.f;
  for (int i = threadIdx.x; i < n; i += 256) s += fabsf(w[i]);
  red[threadIdx.x] = s;
  __syncthreads();
  for (int st = 128; st > 0; st >>= 1) {
    if (threadIdx.x < st) red[threadIdx.x] += red[threadIdx.x + st];
    __syncthreads();
  }
  if (threadIdx.x == 0) scale_out[0] = 1.0f / fmaxf(red[0] / (float)n, 1e-5f);
}

__global__ void k_ternarize(const float* __restrict__ w, const float* __restrict__ scale,
                            bf16* __restrict__ out, int n) {
  const float sc = scale[0], inv = 1.0f / sc;
  for (int i = blockIdx.x * blockDim.x + threadIdx.x; i < n; i += gridDim.x * blockDim.x) {
    float q = rintf(w[i] * sc);
    q = fminf(fmaxf(q, -1.f), 1.f);
    out[i] = (bf16)(q * inv);
  }
}

__global__ void k_act_quant(const float* __restrict__ in, bf16* __restrict__ out, int rowlen) {
  __shared__ float red[256];
  const int row = blockIdx.x;
  const float* rp = in + (size_t)row * rowlen;
  float amax = 0.f;
  for (int c = threadIdx.x; c < rowlen; c += 256) amax = fmaxf(amax, fabsf(rp[c]));
  red[threadIdx.x] = amax;
  __syncthreads();
  for (int st = 128; st > 0; st >>= 1) {
    if (threadIdx.x < st) red[threadIdx.x] = fmaxf(red[threadIdx.x], red[threadIdx.x + st]);
    __syncthreads();
  }
  const float sc = 127.0f / fmaxf(red[0], 1e-5f);
  const float inv = 1.0f / sc;
  bf16* op = out + (size_t)row * rowlen;
  for (int c = threadIdx.x; c < rowlen; c += 256) {
    float q = rintf(rp[c] * sc);
    q = fminf(fmaxf(q, -128.f), 127.f);
    op[c] = (bf16)(q * inv);
  }
}

// conv+proj fold: weff[t*D+i] = sum_o proj_w[o] * conv_w[o,i,t]
__global__ void k_weff(const float* __restrict__ conv_w, const float* __restrict__ proj_w,
                       float* __restrict__ weff) {
  int i = blockIdx.x * blockDim.x + threadIdx.x;
  if (i >= kD) return;
  float a0 = 0.f, a1 = 0.f, a2 = 0.f;
  for (int o = 0; o < kD; ++o) {
    const float pw = proj_w[o];
    const float* cw = conv_w + ((size_t)o * kD + i) * 3;
    a0 += pw * cw[0];
    a1 += pw * cw[1];
    a2 += pw * cw[2];
  }
  weff[0 * kD + i] = a0;
  weff[1 * kD + i] = a1;
  weff[2 * kD + i] = a2;
}

__global__ void k_bias(const float* __restrict__ conv_b, const float* __restrict__ proj_w,
                       float* __restrict__ out) {
  __shared__ float red[256];
  float s = 0.f;
  for (int i = threadIdx.x; i < kD; i += 256) s += proj_w[i] * conv_b[i];
  red[threadIdx.x] = s;
  __syncthreads();
  for (int st = 128; st > 0; st >>= 1) {
    if (threadIdx.x < st) red[threadIdx.x] += red[threadIdx.x + st];
    __syncthreads();
  }
  if (threadIdx.x == 0) out[0] = red[0];
}

__global__ void k_lam(const float* __restrict__ x, const float* __restrict__ weff,
                      const float* __restrict__ bias, const float* __restrict__ proj_b,
                      float* __restrict__ lam) {
  __shared__ float red[256];
  const int s = blockIdx.x;
  float acc = 0.f;
  for (int idx = threadIdx.x; idx < 3 * kD; idx += 256) {
    int t = idx >> 10, i = idx & (kD - 1);
    int sp = s + t - 1;
    if (sp >= 0 && sp < kS) acc += weff[idx] * x[(size_t)sp * kD + i];
  }
  red[threadIdx.x] = acc;
  __syncthreads();
  for (int st = 128; st > 0; st >>= 1) {
    if (threadIdx.x < st) red[threadIdx.x] += red[threadIdx.x + st];
    __syncthreads();
  }
  if (threadIdx.x == 0) {
    float z = red[0] + bias[0] + proj_b[0];
    lam[s] = 1.0f / (1.0f + expf(-z));
  }
}

__global__ void k_rope_bf16(const float* __restrict__ in, const float* __restrict__ cs,
                            const float* __restrict__ sn, bf16* __restrict__ out) {
  int idx = blockIdx.x * blockDim.x + threadIdx.x;  // over kS*kH*32
  if (idx >= kS * kH * 32) return;
  int i = idx & 31;
  int h = (idx >> 5) & (kH - 1);
  int s = idx >> 9;
  float c = cs[s * 32 + i], sv = sn[s * 32 + i];
  size_t base = (size_t)s * kHD + h * kDH + 2 * i;
  float x0 = in[base], x1 = in[base + 1];
  out[base]     = (bf16)(x0 * c - x1 * sv);
  out[base + 1] = (bf16)(x0 * sv + x1 * c);
}

__global__ void k_transpose_v(const float* __restrict__ v, bf16* __restrict__ vt) {
  int idx = blockIdx.x * blockDim.x + threadIdx.x;  // over kS*kHD
  if (idx >= kS * kHD) return;
  int s = idx >> 10;
  int j = idx & (kHD - 1);
  vt[(size_t)j * kS + s] = (bf16)v[idx];
}

// ---- causal flash attention, one 16-row Q tile per wave, 32 keys per step ----
__global__ __launch_bounds__(128) void k_attention(const bf16* __restrict__ Q,
                                                   const bf16* __restrict__ Kb,
                                                   const bf16* __restrict__ Vt,
                                                   float* __restrict__ O) {
  __shared__ __align__(16) bf16 plds[4][16 * 32];
  const int wave = threadIdx.x >> 5, lane = threadIdx.x & 31;
  const int h = blockIdx.y;
  const int qt = blockIdx.x * 4 + wave;
  const int qbase = qt * 16;
  const int n = lane & 15, hi = lane >> 4;

  const bf16* Qp = Q + (size_t)qbase * kHD + h * kDH;
  const v16bf qf0 = load_fragA(Qp, kHD, lane);       // head dims 0..31
  const v16bf qf1 = load_fragA(Qp + 32, kHD, lane);  // head dims 32..63

  v8f o0 = {}, o1 = {}, o2 = {}, o3 = {};
  float mrun[8], lrun[8];
#pragma unroll
  for (int r = 0; r < 8; ++r) { mrun[r] = -1e30f; lrun[r] = 0.f; }

  const int nkb = (qbase + 16 + 31) / 32;
  bf16* pl = &plds[wave][0];
  const bf16* Vp0 = Vt + (size_t)h * kDH * kS;

  for (int kb = 0; kb < nkb; ++kb) {
    const int k0 = kb * 32;
    const bf16* Kp = Kb + (size_t)k0 * kHD + h * kDH;
    v8f s0 = {}, s1 = {};
    {
      v16bf b = load_fragB(Kp, kHD, lane);                  // keys k0..k0+15, dims 0..31
      s0 = WMMA_BF16(qf0, b, s0);
      b = load_fragB(Kp + 32, kHD, lane);                   // dims 32..63
      s0 = WMMA_BF16(qf1, b, s0);
      b = load_fragB(Kp + (size_t)16 * kHD, kHD, lane);     // keys k0+16..k0+31
      s1 = WMMA_BF16(qf0, b, s1);
      b = load_fragB(Kp + (size_t)16 * kHD + 32, kHD, lane);
      s1 = WMMA_BF16(qf1, b, s1);
    }
#pragma unroll
    for (int r = 0; r < 8; ++r) {
      const int q = qbase + hi * 8 + r;
      s0[r] = (k0 + n      <= q) ? s0[r] * 0.125f : -1e30f;
      s1[r] = (k0 + 16 + n <= q) ? s1[r] * 0.125f : -1e30f;
      float mx = fmaxf(s0[r], s1[r]);
#pragma unroll
      for (int msk = 1; msk < 16; msk <<= 1) mx = fmaxf(mx, __shfl_xor(mx, msk, 32));
      const float mnew = fmaxf(mrun[r], mx);
      const float alpha = __expf(mrun[r] - mnew);
      mrun[r] = mnew;
      const float p0 = __expf(s0[r] - mnew);
      const float p1 = __expf(s1[r] - mnew);
      float ps = p0 + p1;
#pragma unroll
      for (int msk = 1; msk < 16; msk <<= 1) ps += __shfl_xor(ps, msk, 32);
      lrun[r] = lrun[r] * alpha + ps;
      pl[(hi * 8 + r) * 32 + n] = (bf16)p0;
      pl[(hi * 8 + r) * 32 + 16 + n] = (bf16)p1;
      o0[r] *= alpha; o1[r] *= alpha; o2[r] *= alpha; o3[r] *= alpha;
    }
    // per-wave LDS RAW: LDS ops from one wave complete in order; fence the compiler
    __builtin_amdgcn_wave_barrier();
    asm volatile("s_wait_dscnt 0x0" ::: "memory");
    const v16bf pf = load_fragA(pl, 32, lane);              // P tile 16x32 as A
    const bf16* Vp = Vp0 + k0;                               // Vt[dim][seq]
    o0 = WMMA_BF16(pf, load_fragB(Vp + (size_t)0  * kS, kS, lane), o0);
    o1 = WMMA_BF16(pf, load_fragB(Vp + (size_t)16 * kS, kS, lane), o1);
    o2 = WMMA_BF16(pf, load_fragB(Vp + (size_t)32 * kS, kS, lane), o2);
    o3 = WMMA_BF16(pf, load_fragB(Vp + (size_t)48 * kS, kS, lane), o3);
    __builtin_amdgcn_wave_barrier();
  }
#pragma unroll
  for (int r = 0; r < 8; ++r) {
    const float inv = 1.0f / lrun[r];
    float* op = O + (size_t)(qbase + hi * 8 + r) * kHD + h * kDH + n;
    op[0]  = o0[r] * inv;
    op[16] = o1[r] * inv;
    op[32] = o2[r] * inv;
    op[48] = o3[r] * inv;
  }
}

__global__ void k_combine(const float* __restrict__ osig, const float* __restrict__ onoise,
                          const float* __restrict__ lam, float* __restrict__ out) {
  int idx = blockIdx.x * blockDim.x + threadIdx.x;
  if (idx >= kS * kHD) return;
  int s = idx >> 10;
  out[idx] = osig[idx] - lam[s] * onoise[idx];
}

// ---- orchestration -----------------------------------------------------------
extern "C" void kernel_launch(void* const* d_in, const int* in_sizes, int n_in,
                              void* d_out, int out_size, void* d_ws, size_t ws_size,
                              hipStream_t stream) {
  (void)in_sizes; (void)n_in; (void)out_size; (void)ws_size;
  const float* x         = (const float*)d_in[0];
  const float* fcos      = (const float*)d_in[1];
  const float* fsin      = (const float*)d_in[2];
  const float* w_down    = (const float*)d_in[3];
  const float* w_u_sig   = (const float*)d_in[4];
  const float* w_u_noise = (const float*)d_in[5];
  const float* w_q_sig   = (const float*)d_in[6];
  const float* w_q_noise = (const float*)d_in[7];
  const float* w_u_val   = (const float*)d_in[8];
  const float* w_o       = (const float*)d_in[9];
  const float* conv_w    = (const float*)d_in[10];
  const float* conv_b    = (const float*)d_in[11];
  const float* proj_w    = (const float*)d_in[12];
  const float* proj_b    = (const float*)d_in[13];

  char* p = (char*)d_ws;
  auto alloc = [&](size_t bytes) -> char* {
    char* r = p;
    p += (bytes + 255) & ~(size_t)255;
    return r;
  };
  float* lam    = (float*)alloc((size_t)kS * 4);
  float* weff   = (float*)alloc((size_t)3 * kD * 4);
  float* scal   = (float*)alloc(64 * 4);
  bf16* xbf     = (bf16*)alloc((size_t)kS * kD * 2);
  bf16* xq      = (bf16*)alloc((size_t)kS * kD * 2);
  bf16* wdownb  = (bf16*)alloc((size_t)kDL * kD * 2);
  bf16* wqusig  = (bf16*)alloc((size_t)kHD * kDL * 2);
  bf16* wqunoi  = (bf16*)alloc((size_t)kHD * kDL * 2);
  bf16* wquval  = (bf16*)alloc((size_t)kHD * kDL * 2);
  bf16* wqqsig  = (bf16*)alloc((size_t)kHD * kD * 2);
  bf16* wqqnoi  = (bf16*)alloc((size_t)kHD * kD * 2);
  bf16* wqo     = (bf16*)alloc((size_t)kD * kHD * 2);
  float* ckv    = (float*)alloc((size_t)kS * kDL * 4);
  bf16* ckvq    = (bf16*)alloc((size_t)kS * kDL * 2);
  float* qsig   = (float*)alloc((size_t)kS * kHD * 4);
  float* qnoi   = (float*)alloc((size_t)kS * kHD * 4);
  float* ksig   = (float*)alloc((size_t)kS * kHD * 4);
  float* knoi   = (float*)alloc((size_t)kS * kHD * 4);
  float* vf     = (float*)alloc((size_t)kS * kHD * 4);
  bf16* qsigb   = (bf16*)alloc((size_t)kS * kHD * 2);
  bf16* qnoib   = (bf16*)alloc((size_t)kS * kHD * 2);
  bf16* ksigb   = (bf16*)alloc((size_t)kS * kHD * 2);
  bf16* knoib   = (bf16*)alloc((size_t)kS * kHD * 2);
  bf16* vtb     = (bf16*)alloc((size_t)kHD * kS * 2);
  bf16* attnqb  = (bf16*)alloc((size_t)kS * kHD * 2);
  float* osig   = ksig;   // ksig (f32) dead after RoPE->bf16
  float* onoi   = knoi;   // knoi (f32) dead after convert->bf16
  float* attnf  = qsig;   // qsig (f32) dead after RoPE->bf16

  // plain bf16 copies for the non-quantized c_kv GEMM
  k_f32_to_bf16<<<1024, 256, 0, stream>>>(x, xbf, kS * kD);
  k_f32_to_bf16<<<256, 256, 0, stream>>>(w_down, wdownb, kDL * kD);

  // weight fake-quant (ternary) -> bf16
  k_wabs_scale<<<1, 256, 0, stream>>>(w_u_sig, kHD * kDL, scal + 1);
  k_ternarize<<<64, 256, 0, stream>>>(w_u_sig, scal + 1, wqusig, kHD * kDL);
  k_wabs_scale<<<1, 256, 0, stream>>>(w_u_noise, kHD * kDL, scal + 2);
  k_ternarize<<<64, 256, 0, stream>>>(w_u_noise, scal + 2, wqunoi, kHD * kDL);
  k_wabs_scale<<<1, 256, 0, stream>>>(w_u_val, kHD * kDL, scal + 3);
  k_ternarize<<<64, 256, 0, stream>>>(w_u_val, scal + 3, wquval, kHD * kDL);
  k_wabs_scale<<<1, 256, 0, stream>>>(w_q_sig, kHD * kD, scal + 4);
  k_ternarize<<<1024, 256, 0, stream>>>(w_q_sig, scal + 4, wqqsig, kHD * kD);
  k_wabs_scale<<<1, 256, 0, stream>>>(w_q_noise, kHD * kD, scal + 5);
  k_ternarize<<<1024, 256, 0, stream>>>(w_q_noise, scal + 5, wqqnoi, kHD * kD);
  k_wabs_scale<<<1, 256, 0, stream>>>(w_o, kD * kHD, scal + 6);
  k_ternarize<<<1024, 256, 0, stream>>>(w_o, scal + 6, wqo, kD * kHD);

  // activation fake-quant of x
  k_act_quant<<<kS, 256, 0, stream>>>(x, xq, kD);

  // conv ∘ proj folded into one mat-vec
  k_weff<<<4, 256, 0, stream>>>(conv_w, proj_w, weff);
  k_bias<<<1, 256, 0, stream>>>(conv_b, proj_w, scal + 0);
  k_lam<<<kS, 256, 0, stream>>>(x, weff, scal + 0, proj_b, lam);

  dim3 blk(256);
  // c_kv = x @ w_down^T  (plain, not bitlinear)
  k_gemm<<<dim3(1, kS / 128), blk, 0, stream>>>(xbf, wdownb, ckv, kS, kDL, kD);
  k_act_quant<<<kS, 256, 0, stream>>>(ckv, ckvq, kDL);

  dim3 g2(kHD / 64, kS / 128);
  k_gemm<<<g2, blk, 0, stream>>>(ckvq, wqusig, ksig, kS, kHD, kDL);
  k_gemm<<<g2, blk, 0, stream>>>(ckvq, wqunoi, knoi, kS, kHD, kDL);
  k_gemm<<<g2, blk, 0, stream>>>(ckvq, wquval, vf,   kS, kHD, kDL);
  k_gemm<<<g2, blk, 0, stream>>>(xq, wqqsig, qsig, kS, kHD, kD);
  k_gemm<<<g2, blk, 0, stream>>>(xq, wqqnoi, qnoi, kS, kHD, kD);

  // RoPE (sig streams) / plain convert (noise, V) to bf16
  k_rope_bf16<<<(kS * kH * 32) / 256, 256, 0, stream>>>(qsig, fcos, fsin, qsigb);
  k_rope_bf16<<<(kS * kH * 32) / 256, 256, 0, stream>>>(ksig, fcos, fsin, ksigb);
  k_f32_to_bf16<<<1024, 256, 0, stream>>>(qnoi, qnoib, kS * kHD);
  k_f32_to_bf16<<<1024, 256, 0, stream>>>(knoi, knoib, kS * kHD);
  k_transpose_v<<<(kS * kHD) / 256, 256, 0, stream>>>(vf, vtb);

  // two causal flash attentions sharing V
  dim3 ga(kS / 16 / 4, kH);
  k_attention<<<ga, 128, 0, stream>>>(qsigb, ksigb, vtb, osig);
  k_attention<<<ga, 128, 0, stream>>>(qnoib, knoib, vtb, onoi);

  // O = O_sig - lam[q] * O_noise, then final bitlinear with w_o -> d_out (f32)
  k_combine<<<(kS * kHD) / 256, 256, 0, stream>>>(osig, onoi, lam, attnf);
  k_act_quant<<<kS, 256, 0, stream>>>(attnf, attnqb, kHD);
  k_gemm<<<g2, blk, 0, stream>>>(attnqb, wqo, (float*)d_out, kS, kHD, kHD);
}
